// PartialEncoderEDDIATSE_6846177870201
// MI455X (gfx1250) — matrix operand order repaired and checked by reading, not compiled, not run on hardware
//
#include <hip/hip_runtime.h>

// ---- problem constants (fixed by setup_inputs) ----
#define B_   128
#define J_   16000
#define D_   32
#define AE_  16
#define H_   64
#define EH_  256
#define A_   2000
#define JPW  25              // junctions per workgroup
#define NBJ  (J_ / JPW)      // 640 j-tiles (exact)
#define EPS_ 1e-5f

typedef __attribute__((ext_vector_type(16))) _Float16 v16h;
typedef __attribute__((ext_vector_type(8)))  float    v8f;
typedef __attribute__((ext_vector_type(2)))  _Float16 h2v;
typedef __attribute__((ext_vector_type(4)))  float    f4v;

// ---------------------------------------------------------------------------
// Kernel 1: per_j[j][h] = b1[h] + feat[j]@W1[1:33] + atse[idx[j]]@W1[33:49]
// ---------------------------------------------------------------------------
__global__ void perj_kernel(const float* __restrict__ feat,
                            const float* __restrict__ atse,
                            const int*   __restrict__ idx,
                            const float* __restrict__ W1,
                            const float* __restrict__ b1,
                            float* __restrict__ per_j) {
  int t = blockIdx.x * blockDim.x + threadIdx.x;
  if (t >= J_ * H_) return;
  int j = t >> 6;
  int h = t & 63;
  const float* fe = feat + (size_t)j * D_;
  const float* ae = atse + (size_t)idx[j] * AE_;
  float s = b1[h];
#pragma unroll 8
  for (int k = 0; k < D_; ++k)  s = fmaf(fe[k], W1[(size_t)(1 + k) * H_ + h], s);
#pragma unroll 8
  for (int k = 0; k < AE_; ++k) s = fmaf(ae[k], W1[(size_t)(1 + D_ + k) * H_ + h], s);
  per_j[t] = s;
}

// ---------------------------------------------------------------------------
// Kernel 2: cnt[b] = sum_j mask[b,j]
// ---------------------------------------------------------------------------
__global__ __launch_bounds__(256) void cnt_kernel(const int* __restrict__ mask,
                                                  float* __restrict__ cnt) {
  __shared__ float red[256];
  const int b = blockIdx.x, t = threadIdx.x;
  float s = 0.f;
  for (int j = t; j < J_; j += 256) s += (float)mask[(size_t)b * J_ + j];
  red[t] = s; __syncthreads();
  for (int o = 128; o > 0; o >>= 1) {
    if (t < o) red[t] += red[t + o];
    __syncthreads();
  }
  if (t == 0) cnt[b] = red[0];
}

// ---------------------------------------------------------------------------
// Kernel 3 (main): fused h-layer + WMMA GEMM + LN + ReLU + masked pooling.
// One wave32 per (16 batch rows x JPW junctions). Lane: g=lane>>4, m=lane&15.
// ---------------------------------------------------------------------------
__global__ __launch_bounds__(32) void fused_kernel(
    const float* __restrict__ x,    const int*   __restrict__ mask,
    const float* __restrict__ per_j,const float* __restrict__ W1,
    const float* __restrict__ g1,   const float* __restrict__ be1,
    const float* __restrict__ W2,   const float* __restrict__ b2,
    const float* __restrict__ g2,   const float* __restrict__ be2,
    float* __restrict__ partial) {
  __shared__ _Float16 ldsA[16][64];   // post-LN/ReLU activations (f16)
  __shared__ _Float16 ldsB[32][64];   // W2 transposed [n][k] (f16)
  __shared__ float    ldsC[32][16];   // D tile transposed [n][row] (f32)

  const int lane = threadIdx.x;
  const int g = lane >> 4;            // k-half / n-half group
  const int m = lane & 15;
  const int b0 = blockIdx.y * 16;
  const int j0 = blockIdx.x * JPW;

  // stage W2^T into LDS as f16
  for (int t = lane; t < H_ * D_; t += 32) {
    int k = t >> 5, n = t & 31;
    ldsB[n][k] = (_Float16)W2[k * D_ + n];
  }
  __syncthreads();

  // loop-invariant B fragments: VGPR v holds K = kh*32 + g*16 + {2v,2v+1},
  // N = nt*16 + m  (contiguous K order, same as A; K in 16-row blocks per lane group)
  v16h bfrag[2][2];
#pragma unroll
  for (int nt = 0; nt < 2; ++nt)
#pragma unroll
    for (int kh = 0; kh < 2; ++kh) {
      union { v16h v; f4v f[2]; } u;
      const _Float16* src = &ldsB[nt * 16 + m][kh * 32 + g * 16];
      u.f[0] = *(const f4v*)(src);
      u.f[1] = *(const f4v*)(src + 8);
      bfrag[nt][kh] = u.v;
    }

  // per-lane parameter slices for k = g*32 + i (h-layer LN #1)
  float w0s[32], g1s[32], be1s[32];
#pragma unroll
  for (int i = 0; i < 32; ++i) {
    int k = g * 32 + i;
    w0s[i]  = W1[k];            // h_W1 row 0
    g1s[i]  = g1[k];
    be1s[i] = be1[k];
  }
  // epilogue params: store-side bias columns (n=m, n=16+m),
  // compute-side LN#2 slices for n = g*16 + i
  const float b2a = b2[m], b2b = b2[16 + m];
  float g2s[16], be2s[16];
#pragma unroll
  for (int i = 0; i < 16; ++i) {
    int n = g * 16 + i;
    g2s[i]  = g2[n];
    be2s[i] = be2[n];
  }

  float acc[16];                // pooled sum: row b0+m, n = g*16 + i
#pragma unroll
  for (int i = 0; i < 16; ++i) acc[i] = 0.f;

  for (int jj = 0; jj < JPW; ++jj) {
    const int j = j0 + jj;
    const float xv = x[(size_t)(b0 + m) * J_ + j];
    const float mk = (float)mask[(size_t)(b0 + m) * J_ + j];
    const f4v* pj = (const f4v*)(per_j + (size_t)j * H_ + g * 32);
    if (jj + 1 < JPW)
      __builtin_prefetch(per_j + (size_t)(j + 1) * H_ + g * 32, 0, 1);

    // h1 slice: 32 values (k = g*32 + i) for row b0+m
    float h[32];
#pragma unroll
    for (int q = 0; q < 8; ++q) {
      f4v p = pj[q];
      h[4*q+0] = fmaf(xv, w0s[4*q+0], p.x);
      h[4*q+1] = fmaf(xv, w0s[4*q+1], p.y);
      h[4*q+2] = fmaf(xv, w0s[4*q+2], p.z);
      h[4*q+3] = fmaf(xv, w0s[4*q+3], p.w);
    }
    // LayerNorm over 64 (pair lanes l and l^16 hold the two halves)
    float s = 0.f;
#pragma unroll
    for (int i = 0; i < 32; ++i) s += h[i];
    s += __shfl_xor(s, 16, 32);
    const float mu = s * (1.0f / 64.0f);
    float vv = 0.f;
#pragma unroll
    for (int i = 0; i < 32; ++i) { float d = h[i] - mu; vv = fmaf(d, d, vv); }
    vv += __shfl_xor(vv, 16, 32);
    const float inv = __frsqrt_rn(vv * (1.0f / 64.0f) + EPS_);
    // normalize + ReLU -> f16 tile in LDS
#pragma unroll
    for (int i = 0; i < 32; i += 2) {
      float a0 = fmaxf(fmaf((h[i]   - mu) * inv, g1s[i],   be1s[i]),   0.f);
      float a1 = fmaxf(fmaf((h[i+1] - mu) * inv, g1s[i+1], be1s[i+1]), 0.f);
      h2v p; p.x = (_Float16)a0; p.y = (_Float16)a1;
      *(h2v*)&ldsA[m][g * 32 + i] = p;
    }
    __syncthreads();

    // A fragments: VGPR 0-7: K = kh*32 + g*16 + {0..15}
    v16h afrag[2];
#pragma unroll
    for (int kh = 0; kh < 2; ++kh) {
      union { v16h v; f4v f[2]; } u;
      const _Float16* src = &ldsA[m][kh * 32 + g * 16];
      u.f[0] = *(const f4v*)(src);
      u.f[1] = *(const f4v*)(src + 8);
      afrag[kh] = u.v;
    }

    // D(16x32 f32) = A(16x64) @ W2(64x32), chained K accumulation
    v8f d0 = {}, d1 = {};
    d0 = __builtin_amdgcn_wmma_f32_16x16x32_f16(false, afrag[0], false, bfrag[0][0], (short)0, d0, false, false);
    d0 = __builtin_amdgcn_wmma_f32_16x16x32_f16(false, afrag[1], false, bfrag[0][1], (short)0, d0, false, false);
    d1 = __builtin_amdgcn_wmma_f32_16x16x32_f16(false, afrag[0], false, bfrag[1][0], (short)0, d1, false, false);
    d1 = __builtin_amdgcn_wmma_f32_16x16x32_f16(false, afrag[1], false, bfrag[1][1], (short)0, d1, false, false);

    // Transpose D tile (+bias) through LDS so each lane owns a row-half:
    // C/D layout: VGPR r, lane -> row M = 8*g + r, col N = m (d0) / 16+m (d1)
    // store ldsC[n][row]: 4x b128, conflict-free on 64 banks
    {
      f4v t;
      t.x = d0[0] + b2a; t.y = d0[1] + b2a; t.z = d0[2] + b2a; t.w = d0[3] + b2a;
      *(f4v*)&ldsC[m][8 * g] = t;
      t.x = d0[4] + b2a; t.y = d0[5] + b2a; t.z = d0[6] + b2a; t.w = d0[7] + b2a;
      *(f4v*)&ldsC[m][8 * g + 4] = t;
      t.x = d1[0] + b2b; t.y = d1[1] + b2b; t.z = d1[2] + b2b; t.w = d1[3] + b2b;
      *(f4v*)&ldsC[16 + m][8 * g] = t;
      t.x = d1[4] + b2b; t.y = d1[5] + b2b; t.z = d1[6] + b2b; t.w = d1[7] + b2b;
      *(f4v*)&ldsC[16 + m][8 * g + 4] = t;
    }
    __syncthreads();

    // lane now handles row = m, n = g*16 + i. LN over 32 with ONE lane-pair
    // exchange (lane^16 holds the other 16 columns of the same row).
    float e[16];
#pragma unroll
    for (int i = 0; i < 16; ++i) e[i] = ldsC[g * 16 + i][m];
    float sr = 0.f;
#pragma unroll
    for (int i = 0; i < 16; ++i) sr += e[i];
    sr += __shfl_xor(sr, 16, 32);
    const float mur = sr * (1.0f / 32.0f);
    float vr = 0.f;
#pragma unroll
    for (int i = 0; i < 16; ++i) { float d = e[i] - mur; vr = fmaf(d, d, vr); }
    vr += __shfl_xor(vr, 16, 32);
    const float invr = __frsqrt_rn(vr * (1.0f / 32.0f) + EPS_);
    // normalize + ReLU + mask-weighted accumulate (mask is lane-local: row m)
#pragma unroll
    for (int i = 0; i < 16; ++i) {
      float o = fmaxf(fmaf((e[i] - mur) * invr, g2s[i], be2s[i]), 0.f);
      acc[i] = fmaf(mk, o, acc[i]);
    }
    __syncthreads();   // protect ldsA/ldsC before next iteration
  }

  // pooled partials: contiguous per lane -> 4x global_store_b128
  float* dst = partial + (size_t)blockIdx.x * (B_ * D_) + (b0 + m) * D_ + g * 16;
#pragma unroll
  for (int q = 0; q < 4; ++q) {
    f4v t;
    t.x = acc[4*q+0]; t.y = acc[4*q+1]; t.z = acc[4*q+2]; t.w = acc[4*q+3];
    *(f4v*)&dst[4*q] = t;
  }
}

// ---------------------------------------------------------------------------
// Kernel 4: c_sum[b][n] = sum over NBJ partial tiles
// ---------------------------------------------------------------------------
__global__ __launch_bounds__(256) void reduce_kernel(const float* __restrict__ partial,
                                                     float* __restrict__ c_sum) {
  int t = blockIdx.x * blockDim.x + threadIdx.x;   // 0..B*D-1
  float s = 0.f;
  for (int p = 0; p < NBJ; ++p) s += partial[(size_t)p * (B_ * D_) + t];
  c_sum[t] = s;
}

// ---------------------------------------------------------------------------
// Kernel 5: masked mean + encoder MLP (32 -> 256 -> 64, LN+ReLU each layer)
// ---------------------------------------------------------------------------
__global__ __launch_bounds__(256) void encoder_kernel(
    const float* __restrict__ c_sum, const float* __restrict__ cnt,
    const float* __restrict__ W1, const float* __restrict__ b1,
    const float* __restrict__ g1, const float* __restrict__ be1,
    const float* __restrict__ W2, const float* __restrict__ b2,
    const float* __restrict__ g2, const float* __restrict__ be2,
    float* __restrict__ out) {
  __shared__ float cs[32];
  __shared__ float red[256];
  __shared__ float e1s[256];
  const int b = blockIdx.x, t = threadIdx.x;
  if (t < 32) {
    float cn = cnt[b];
    cs[t] = (cn > 0.f) ? c_sum[b * D_ + t] / fmaxf(cn, 1.f) : 0.f;
  }
  __syncthreads();

  // layer 1: 32 -> 256
  float v = b1[t];
#pragma unroll 8
  for (int k = 0; k < 32; ++k) v = fmaf(cs[k], W1[k * EH_ + t], v);
  red[t] = v; __syncthreads();
  for (int o = 128; o > 0; o >>= 1) { if (t < o) red[t] += red[t + o]; __syncthreads(); }
  const float mu1 = red[0] * (1.f / 256.f);
  __syncthreads();
  const float d1v = v - mu1;
  red[t] = d1v * d1v; __syncthreads();
  for (int o = 128; o > 0; o >>= 1) { if (t < o) red[t] += red[t + o]; __syncthreads(); }
  const float inv1 = __frsqrt_rn(red[0] * (1.f / 256.f) + EPS_);
  __syncthreads();
  e1s[t] = fmaxf(fmaf(d1v * inv1, g1[t], be1[t]), 0.f);
  __syncthreads();

  // layer 2: 256 -> 64
  float o2 = 0.f;
  if (t < 64) {
    o2 = b2[t];
    for (int k = 0; k < 256; ++k) o2 = fmaf(e1s[k], W2[k * 64 + t], o2);
  }
  red[t] = (t < 64) ? o2 : 0.f; __syncthreads();
  for (int o = 128; o > 0; o >>= 1) { if (t < o) red[t] += red[t + o]; __syncthreads(); }
  const float mu2 = red[0] * (1.f / 64.f);
  __syncthreads();
  const float d2v = o2 - mu2;
  red[t] = (t < 64) ? d2v * d2v : 0.f; __syncthreads();
  for (int o = 128; o > 0; o >>= 1) { if (t < o) red[t] += red[t + o]; __syncthreads(); }
  const float inv2 = __frsqrt_rn(red[0] * (1.f / 64.f) + EPS_);
  __syncthreads();
  if (t < 64) {
    float val = fmaxf(fmaf(d2v * inv2, g2[t], be2[t]), 0.f);
    if (t < 32) out[b * 32 + t] = val;                       // mu
    else        out[B_ * 32 + b * 32 + (t - 32)] = val;      // logvar
  }
}

// ---------------------------------------------------------------------------
extern "C" void kernel_launch(void* const* d_in, const int* in_sizes, int n_in,
                              void* d_out, int out_size, void* d_ws, size_t ws_size,
                              hipStream_t stream) {
  const float* x     = (const float*)d_in[0];
  const int*   mask  = (const int*)d_in[1];
  const int*   aidx  = (const int*)d_in[2];
  const float* feat  = (const float*)d_in[3];
  const float* atse  = (const float*)d_in[4];
  const float* hW1   = (const float*)d_in[5];
  const float* hb1   = (const float*)d_in[6];
  const float* hg1   = (const float*)d_in[7];
  const float* hbe1  = (const float*)d_in[8];
  const float* hW2   = (const float*)d_in[9];
  const float* hb2   = (const float*)d_in[10];
  const float* hg2   = (const float*)d_in[11];
  const float* hbe2  = (const float*)d_in[12];
  const float* eW1   = (const float*)d_in[13];
  const float* eb1   = (const float*)d_in[14];
  const float* eg1   = (const float*)d_in[15];
  const float* ebe1  = (const float*)d_in[16];
  const float* eW2   = (const float*)d_in[17];
  const float* eb2   = (const float*)d_in[18];
  const float* eg2   = (const float*)d_in[19];
  const float* ebe2  = (const float*)d_in[20];
  float* out = (float*)d_out;

  // workspace carve-up (all regions fully overwritten every launch)
  char* ws = (char*)d_ws;
  float* per_j   = (float*)(ws);                                   // J*H
  float* partial = (float*)(ws + (((size_t)J_ * H_ * 4 + 255) & ~(size_t)255));  // NBJ*B*D
  float* cnt     = (float*)((char*)partial + (((size_t)NBJ * B_ * D_ * 4 + 255) & ~(size_t)255));
  float* c_sum   = cnt + 256;                                      // B*D

  perj_kernel<<<(J_ * H_ + 255) / 256, 256, 0, stream>>>(feat, atse, aidx, hW1, hb1, per_j);
  cnt_kernel<<<B_, 256, 0, stream>>>(mask, cnt);
  fused_kernel<<<dim3(NBJ, B_ / 16), 32, 0, stream>>>(
      x, mask, per_j, hW1, hg1, hbe1, hW2, hb2, hg2, hbe2, partial);
  reduce_kernel<<<(B_ * D_) / 256, 256, 0, stream>>>(partial, c_sum);
  encoder_kernel<<<B_, 256, 0, stream>>>(c_sum, cnt, eW1, eb1, eg1, ebe1,
                                         eW2, eb2, eg2, ebe2, out);
}
